// ChronosPnLPolicy_85418309583370
// MI455X (gfx1250) — matrix operand order repaired and checked by reading, not compiled, not run on hardware
//
#include <hip/hip_runtime.h>
#include <hip/hip_bf16.h>
#include <math.h>
#include <cstdint>

typedef unsigned short u16t;
typedef __attribute__((ext_vector_type(16))) __bf16 v16bf;
typedef __attribute__((ext_vector_type(8)))  float  v8f;

#define EPSV 1.1920929e-07f
#define NEGBIG (-1.0e30f)

#define ASYNC_WAIT(n) asm volatile("s_wait_asynccnt %0" :: "i"(n) : "memory")

__device__ __forceinline__ u16t f2bf(float f){
  union { float f; unsigned u; } c; c.f = f;
  unsigned u = c.u;
  unsigned r = u + 0x7FFFu + ((u >> 16) & 1u);
  return (u16t)(r >> 16);
}

__device__ __forceinline__ void async_b128(const void* gaddr, const void* lds_ptr){
  unsigned l = (unsigned)(size_t)lds_ptr;                 // low 32 bits of LDS aperture = LDS offset
  unsigned long long g = (unsigned long long)(size_t)gaddr;
  asm volatile("global_load_async_to_lds_b128 %0, %1, off" :: "v"(l), "v"(g) : "memory");
}

union FragAB { v16bf v; uint4 q[2]; };

// ---------------- f32 -> bf16 conversion ----------------
__global__ void k_cvt_bf16(const float* __restrict__ src, u16t* __restrict__ dst, int n){
  for (int i = blockIdx.x * blockDim.x + threadIdx.x; i < n; i += gridDim.x * blockDim.x)
    dst[i] = f2bf(src[i]);
}

// f32 [K,Nsrc] -> bf16 [K,Ndst] zero-padded columns
__global__ void k_cvt_pad(const float* __restrict__ src, u16t* __restrict__ dst,
                          int K, int Nsrc, int Ndst){
  int n = K * Ndst;
  for (int i = blockIdx.x * blockDim.x + threadIdx.x; i < n; i += gridDim.x * blockDim.x){
    int r = i / Ndst, c = i - r * Ndst;
    dst[i] = (c < Nsrc) ? f2bf(src[r * Nsrc + c]) : (u16t)0;
  }
}

// ---------------- generic bf16 WMMA GEMM ----------------
// C[M,N] = A[M,K] @ W[K,ldw(>=N)]; block tile 128x256, wave tile 64x64, k-step 32,
// double-buffered LDS fed by global_load_async_to_lds_b128 (ASYNCcnt).
#define EPI_F32   0
#define EPI_GELU  1
#define EPI_RELU2 2
#define EPI_RES   3
#define EPI_BF16  4

__device__ __forceinline__ void gemm_load_tiles(const u16t* __restrict__ A, const u16t* __restrict__ W,
                                                int K, int ldw, int m0, int n0, int k0, int tid,
                                                u16t* As, u16t* Ws){
  // A tile: 128 x 32 bf16 = 512 x 16B chunks, 2 per thread
  #pragma unroll
  for (int it = 0; it < 2; ++it){
    int idx = tid + it * 256;
    int r = idx >> 2, c = (idx & 3) * 8;
    async_b128(A + (size_t)(m0 + r) * K + k0 + c, As + r * 40 + c);
  }
  // W tile: 32 x 256 bf16 = 1024 x 16B chunks, 4 per thread
  #pragma unroll
  for (int it = 0; it < 4; ++it){
    int idx = tid + it * 256;
    int r = idx >> 5, c = (idx & 31) * 8;
    async_b128(W + (size_t)(k0 + r) * ldw + n0 + c, Ws + r * 264 + c);
  }
}

__global__ __launch_bounds__(256) void k_gemm(const u16t* __restrict__ A, const u16t* __restrict__ W,
                                              int M, int N, int K, int ldw,
                                              float* __restrict__ Cf, u16t* __restrict__ Cb,
                                              const float* __restrict__ Res, int epi){
  __shared__ u16t As[2][128 * 40];   // pad 32->40: 16B aligned rows, conflict-free frag reads
  __shared__ u16t Ws[2][32 * 264];   // pad 256->264
  const int tid  = threadIdx.x;
  const int lane = tid & 31, wave = tid >> 5;
  const int wm = wave >> 2, wn = wave & 3;       // 2 x 4 wave grid -> 128 x 256 block tile
  const int half = lane >> 4, lm = lane & 15;
  const int m0 = blockIdx.y * 128, n0 = blockIdx.x * 256;

  v8f acc[4][4];
  #pragma unroll
  for (int i = 0; i < 4; ++i)
    #pragma unroll
    for (int j = 0; j < 4; ++j) acc[i][j] = (v8f){0,0,0,0,0,0,0,0};

  const int steps = K >> 5;
  gemm_load_tiles(A, W, K, ldw, m0, n0, 0, tid, As[0], Ws[0]);

  for (int s = 0; s < steps; ++s){
    const int cur = s & 1;
    if (s + 1 < steps){
      gemm_load_tiles(A, W, K, ldw, m0, n0, (s + 1) * 32, tid, As[1 - cur], Ws[1 - cur]);
      ASYNC_WAIT(6);                 // 6 newest outstanding -> current buffer complete
    } else {
      ASYNC_WAIT(0);
    }
    __syncthreads();

    FragAB a[4], b[4];
    #pragma unroll
    for (int i = 0; i < 4; ++i){
      const u16t* p = &As[cur][(wm * 64 + i * 16 + lm) * 40 + half * 8];
      a[i].q[0] = *(const uint4*)p;
      a[i].q[1] = *(const uint4*)(p + 16);
    }
    #pragma unroll
    for (int j = 0; j < 4; ++j){
      const u16t* p = &Ws[cur][wn * 64 + j * 16 + lm];
      u16t t[16];
      #pragma unroll
      for (int e = 0; e < 16; ++e) t[e] = p[((e & 7) + 8 * half + ((e >> 3) << 4)) * 264];
      __builtin_memcpy(&b[j], t, 32);
    }
    #pragma unroll
    for (int i = 0; i < 4; ++i)
      #pragma unroll
      for (int j = 0; j < 4; ++j)
        acc[i][j] = __builtin_amdgcn_wmma_f32_16x16x32_bf16(false, a[i].v, false, b[j].v,
                                                            (short)0, acc[i][j], false, false);
    __syncthreads();                 // all waves done with buffer cur before it is refilled
  }

  #pragma unroll
  for (int i = 0; i < 4; ++i){
    #pragma unroll
    for (int j = 0; j < 4; ++j){
      #pragma unroll
      for (int g = 0; g < 8; ++g){
        int m = m0 + wm * 64 + i * 16 + g + 8 * half;
        int n = n0 + wn * 64 + j * 16 + lm;
        if (n < N){
          float v = acc[i][j][g];
          size_t idx = (size_t)m * N + n;
          if (epi == EPI_F32)       Cf[idx] = v;
          else if (epi == EPI_GELU) Cb[idx] = f2bf(0.5f * v * (1.0f + erff(v * 0.70710678118654752f)));
          else if (epi == EPI_RELU2){ float r = fmaxf(v, 0.f); Cb[idx] = f2bf(r * r); }
          else if (epi == EPI_RES)  Cf[idx] = Res[idx] + v;
          else                      Cb[idx] = f2bf(v);
        }
      }
    }
  }
}

// ---------------- RMSNorm over rows of 1024 ----------------
__global__ __launch_bounds__(256) void k_rmsn(const float* __restrict__ in,
                                              float* __restrict__ outf, u16t* __restrict__ outb){
  __shared__ float sm[8];
  const int row = blockIdx.x, tid = threadIdx.x;
  const float* r = in + (size_t)row * 1024;
  float x[4]; float ss = 0.f;
  #pragma unroll
  for (int i = 0; i < 4; ++i){ x[i] = r[tid + i * 256]; ss += x[i] * x[i]; }
  #pragma unroll
  for (int off = 16; off >= 1; off >>= 1) ss += __shfl_xor(ss, off, 32);
  if ((tid & 31) == 0) sm[tid >> 5] = ss;
  __syncthreads();
  if (tid == 0){ float t = 0; for (int i = 0; i < 8; ++i) t += sm[i]; sm[0] = t; }
  __syncthreads();
  float sc = rsqrtf(sm[0] * (1.0f / 1024.0f) + EPSV);
  #pragma unroll
  for (int i = 0; i < 4; ++i){
    float v = x[i] * sc;
    size_t idx = (size_t)row * 1024 + tid + i * 256;
    if (outf) outf[idx] = v;
    if (outb) outb[idx] = f2bf(v);
  }
}

// ---------------- RoPE tables ----------------
__global__ void k_rope_setup(float* __restrict__ cosb, float* __restrict__ sinb){
  int t = blockIdx.x, i = threadIdx.x;   // 1024 x 32
  float inv = powf(10000.0f, -(float)(2 * i) / 64.0f);
  float f = (float)t * inv;
  cosb[t * 32 + i] = cosf(f);
  sinb[t * 32 + i] = sinf(f);
}

// rope + per-head rmsnorm for Q: qf[B*T,1024] -> qb[(b*16+h)*1024+t][64] bf16
__global__ void k_rope_q(const float* __restrict__ qf, u16t* __restrict__ qb,
                         const float* __restrict__ cosb, const float* __restrict__ sinb){
  int t = blockIdx.x, h = blockIdx.y, b = blockIdx.z, i = threadIdx.x;  // 32 threads = 32 pairs
  const float* src = qf + ((size_t)(b * 1024 + t)) * 1024 + h * 64;
  float xr = src[2 * i], xi = src[2 * i + 1];
  float c = cosb[t * 32 + i], s = sinb[t * 32 + i];
  float orr = xr * c - xi * s, oi = xr * s + xi * c;
  float ss = orr * orr + oi * oi;
  #pragma unroll
  for (int off = 16; off >= 1; off >>= 1) ss += __shfl_xor(ss, off, 32);
  float sc = rsqrtf(ss * (1.0f / 64.0f) + EPSV);
  u16t* dst = qb + (((size_t)(b * 16 + h)) * 1024 + t) * 64;
  dst[2 * i]     = f2bf(orr * sc);
  dst[2 * i + 1] = f2bf(oi * sc);
}

__global__ void k_rope_k(const float* __restrict__ kf, u16t* __restrict__ kb,
                         const float* __restrict__ cosb, const float* __restrict__ sinb){
  int t = blockIdx.x, b = blockIdx.y, i = threadIdx.x;
  const float* src = kf + ((size_t)(b * 1024 + t)) * 64;
  float xr = src[2 * i], xi = src[2 * i + 1];
  float c = cosb[t * 32 + i], s = sinb[t * 32 + i];
  float orr = xr * c - xi * s, oi = xr * s + xi * c;
  float ss = orr * orr + oi * oi;
  #pragma unroll
  for (int off = 16; off >= 1; off >>= 1) ss += __shfl_xor(ss, off, 32);
  float sc = rsqrtf(ss * (1.0f / 64.0f) + EPSV);
  u16t* dst = kb + ((size_t)(b * 1024 + t)) * 64;
  dst[2 * i]     = f2bf(orr * sc);
  dst[2 * i + 1] = f2bf(oi * sc);
}

// ---------------- WMMA flash attention ----------------
// grid (T/128, H=16, B=4), 256 threads (8 waves x 16 q-rows). Causal, GQA n_kv=1.
__global__ __launch_bounds__(256) void k_flash(const u16t* __restrict__ Q, const u16t* __restrict__ Kb,
                                               const u16t* __restrict__ Vb, u16t* __restrict__ Y){
  __shared__ u16t Kt[64 * 72];    // [d][k-row] (transposed K tile)
  __shared__ u16t Vs[64 * 72];    // [k-row][d], async-loaded
  __shared__ u16t Pl[128 * 72];   // per-wave P scratch (16 rows each)
  const int tid = threadIdx.x, lane = tid & 31, wave = tid >> 5;
  const int half = lane >> 4, lm = lane & 15;
  const int q0 = blockIdx.x * 128, hh = blockIdx.y, b = blockIdx.z;

  FragAB qa[2];
  {
    const u16t* qrow = Q + (((size_t)(b * 16 + hh)) * 1024 + q0 + wave * 16 + lm) * 64;
    #pragma unroll
    for (int ds = 0; ds < 2; ++ds){
      qa[ds].q[0] = *(const uint4*)(qrow + ds * 32 + half * 8);
      qa[ds].q[1] = *(const uint4*)(qrow + ds * 32 + half * 8 + 16);
    }
  }

  float mi[8], li[8];
  v8f o[4];
  #pragma unroll
  for (int g = 0; g < 8; ++g){ mi[g] = NEGBIG; li[g] = 0.f; }
  #pragma unroll
  for (int nf = 0; nf < 4; ++nf) o[nf] = (v8f){0,0,0,0,0,0,0,0};

  const int nkt = q0 / 64 + 2;    // causal: k-tiles covering k <= q0+127
  for (int kt = 0; kt < nkt; ++kt){
    __syncthreads();
    const int kbase = kt * 64;
    // V tile: async DMA into LDS (64 x 128B)
    #pragma unroll
    for (int it = 0; it < 2; ++it){
      int idx = tid + it * 256;
      int r = idx >> 3, c = (idx & 7) * 8;
      async_b128(Vb + ((size_t)(b * 1024) + kbase + r) * 64 + c, &Vs[r * 72 + c]);
    }
    // K tile: transposed into LDS (element-wise)
    for (int idx = tid; idx < 64 * 64; idx += 256){
      int r = idx >> 6, d = idx & 63;
      Kt[d * 72 + r] = Kb[((size_t)(b * 1024) + kbase + r) * 64 + d];
    }
    ASYNC_WAIT(0);
    __syncthreads();

    // S = Q K^T  (16 q rows x 64 k rows per wave)
    v8f s[4];
    #pragma unroll
    for (int nf = 0; nf < 4; ++nf) s[nf] = (v8f){0,0,0,0,0,0,0,0};
    #pragma unroll
    for (int ds = 0; ds < 2; ++ds){
      #pragma unroll
      for (int nf = 0; nf < 4; ++nf){
        FragAB kfr;
        const u16t* p = &Kt[(ds * 32) * 72 + nf * 16 + lm];
        u16t t[16];
        #pragma unroll
        for (int e = 0; e < 16; ++e) t[e] = p[((e & 7) + 8 * half + ((e >> 3) << 4)) * 72];
        __builtin_memcpy(&kfr, t, 32);
        s[nf] = __builtin_amdgcn_wmma_f32_16x16x32_bf16(false, qa[ds].v, false, kfr.v,
                                                        (short)0, s[nf], false, false);
      }
    }
    // scale + causal mask + row max
    float rmax[8];
    #pragma unroll
    for (int g = 0; g < 8; ++g) rmax[g] = NEGBIG;
    #pragma unroll
    for (int nf = 0; nf < 4; ++nf){
      int kcol = kbase + nf * 16 + lm;
      #pragma unroll
      for (int g = 0; g < 8; ++g){
        int qrow = q0 + wave * 16 + g + 8 * half;
        float sv = (kcol <= qrow) ? s[nf][g] * 0.125f : NEGBIG;
        s[nf][g] = sv;
        rmax[g] = fmaxf(rmax[g], sv);
      }
    }
    #pragma unroll
    for (int g = 0; g < 8; ++g)
      #pragma unroll
      for (int off = 1; off < 16; off <<= 1) rmax[g] = fmaxf(rmax[g], __shfl_xor(rmax[g], off, 32));

    float alpha[8], rsum[8], mnew[8];
    #pragma unroll
    for (int g = 0; g < 8; ++g){
      mnew[g]  = fmaxf(mi[g], rmax[g]);
      alpha[g] = __expf(mi[g] - mnew[g]);
      mi[g]    = mnew[g];
      rsum[g]  = 0.f;
    }
    // P = exp(S - mnew), requantize bf16 through per-wave LDS
    #pragma unroll
    for (int nf = 0; nf < 4; ++nf){
      #pragma unroll
      for (int g = 0; g < 8; ++g){
        float pv = __expf(s[nf][g] - mnew[g]);
        rsum[g] += pv;
        Pl[(wave * 16 + g + 8 * half) * 72 + nf * 16 + lm] = f2bf(pv);
      }
    }
    #pragma unroll
    for (int g = 0; g < 8; ++g){
      #pragma unroll
      for (int off = 1; off < 16; off <<= 1) rsum[g] += __shfl_xor(rsum[g], off, 32);
      li[g] = li[g] * alpha[g] + rsum[g];
    }
    asm volatile("s_wait_dscnt 0" ::: "memory");  // P writes visible before A-frag reload
    // O = O*alpha + P @ V
    #pragma unroll
    for (int nf = 0; nf < 4; ++nf)
      #pragma unroll
      for (int g = 0; g < 8; ++g) o[nf][g] *= alpha[g];
    #pragma unroll
    for (int ks = 0; ks < 2; ++ks){
      FragAB pa;
      const u16t* pp = &Pl[(wave * 16 + lm) * 72 + ks * 32 + half * 8];
      pa.q[0] = *(const uint4*)pp;
      pa.q[1] = *(const uint4*)(pp + 16);
      #pragma unroll
      for (int nf = 0; nf < 4; ++nf){
        FragAB vf;
        const u16t* vp = &Vs[(ks * 32) * 72 + nf * 16 + lm];
        u16t t[16];
        #pragma unroll
        for (int e = 0; e < 16; ++e) t[e] = vp[((e & 7) + 8 * half + ((e >> 3) << 4)) * 72];
        __builtin_memcpy(&vf, t, 32);
        o[nf] = __builtin_amdgcn_wmma_f32_16x16x32_bf16(false, pa.v, false, vf.v,
                                                        (short)0, o[nf], false, false);
      }
    }
  }
  // epilogue: O / l, write y[(b*T+q)][h*64+d] bf16
  #pragma unroll
  for (int nf = 0; nf < 4; ++nf){
    #pragma unroll
    for (int g = 0; g < 8; ++g){
      int q = q0 + wave * 16 + g + 8 * half;
      float inv = 1.0f / li[g];
      Y[((size_t)(b * 1024) + q) * 1024 + hh * 64 + nf * 16 + lm] = f2bf(o[nf][g] * inv);
    }
  }
}

// ---------------- PnL encoder (tiny) ----------------
__device__ float blk_reduce(float v, float* sm){
  #pragma unroll
  for (int off = 16; off >= 1; off >>= 1) v += __shfl_xor(v, off, 32);
  if ((threadIdx.x & 31) == 0) sm[threadIdx.x >> 5] = v;
  __syncthreads();
  if (threadIdx.x == 0){ float t = 0; for (int i = 0; i < 8; ++i) t += sm[i]; sm[0] = t; }
  __syncthreads();
  float r = sm[0];
  __syncthreads();
  return r;
}

__global__ __launch_bounds__(256) void k_pnl_stats(const float* __restrict__ pnl, float* __restrict__ stats){
  __shared__ float sm[8];
  int b = blockIdx.x, tid = threadIdx.x;
  const float* p = pnl + b * 210;
  float s = 0, s2 = 0, rec = 0, old = 0;
  for (int i = tid; i < 210; i += 256){
    float v = p[i];
    s += v; s2 += v * v;
    if (i >= 203) rec += v;
    if (i >= 175 && i < 203) old += v;
  }
  s   = blk_reduce(s, sm);
  s2  = blk_reduce(s2, sm);
  rec = blk_reduce(rec, sm);
  old = blk_reduce(old, sm);
  if (tid == 0){
    float mean = s / 210.f;
    float var  = (s2 - s * s / 210.f) / 209.f;
    float sd   = sqrtf(fmaxf(var, 0.f)) + 1e-8f;
    stats[b * 5 + 0] = mean;
    stats[b * 5 + 1] = sd;
    stats[b * 5 + 2] = s;
    stats[b * 5 + 3] = rec / 7.f;
    stats[b * 5 + 4] = rec / 7.f - old / 28.f;
  }
}

__global__ void k_pnl1(const float* __restrict__ pnl, const float* __restrict__ w1, float* __restrict__ t1){
  int b = blockIdx.x, j = blockIdx.y * 256 + threadIdx.x;
  float acc = 0;
  for (int i = 0; i < 210; ++i) acc += pnl[b * 210 + i] * w1[i * 1024 + j];
  t1[b * 1024 + j] = 0.5f * acc * (1.0f + erff(acc * 0.70710678118654752f));
}

__global__ void k_pnl2(const float* __restrict__ t1, const float* __restrict__ w2,
                       const float* __restrict__ stats, const float* __restrict__ sw,
                       float* __restrict__ h){
  int b = blockIdx.x, j = blockIdx.y * 256 + threadIdx.x;
  float acc = 0;
  for (int i = 0; i < 1024; ++i) acc += t1[b * 1024 + i] * w2[i * 1024 + j];
  for (int s = 0; s < 5; ++s)    acc += stats[b * 5 + s] * sw[s * 1024 + j];
  h[((size_t)(b * 1024 + 1023)) * 1024 + j] += acc;
}

// ---------------- final rmsnorm + 4 heads ----------------
__global__ __launch_bounds__(256) void k_heads(const float* __restrict__ h,
                                               const float* __restrict__ wl, const float* __restrict__ wsz,
                                               const float* __restrict__ wby, const float* __restrict__ wsl,
                                               float* __restrict__ out){
  __shared__ float sm[8];
  __shared__ float pooled[1024];
  int b = blockIdx.x, tid = threadIdx.x;
  const float* row = h + ((size_t)(b * 1024 + 1023)) * 1024;
  float x[4]; float ss = 0;
  #pragma unroll
  for (int i = 0; i < 4; ++i){ x[i] = row[tid + i * 256]; ss += x[i] * x[i]; }
  float total = blk_reduce(ss, sm);
  float sc = rsqrtf(total * (1.0f / 1024.0f) + EPSV);
  #pragma unroll
  for (int i = 0; i < 4; ++i) pooled[tid + i * 256] = x[i] * sc;
  __syncthreads();
  for (int j = 0; j < 8; ++j){
    float p = 0;
    #pragma unroll
    for (int i = 0; i < 4; ++i){ int c = tid + i * 256; p += pooled[c] * wl[c * 8 + j]; }
    p = blk_reduce(p, sm);
    if (tid == 0) out[b * 8 + j] = p;
  }
  float p1 = 0, p2 = 0, p3 = 0;
  #pragma unroll
  for (int i = 0; i < 4; ++i){
    int c = tid + i * 256;
    p1 += pooled[c] * wsz[c];
    p2 += pooled[c] * wby[c];
    p3 += pooled[c] * wsl[c];
  }
  p1 = blk_reduce(p1, sm); if (tid == 0) out[32 + b] = p1;
  p2 = blk_reduce(p2, sm); if (tid == 0) out[36 + b] = p2;
  p3 = blk_reduce(p3, sm); if (tid == 0) out[40 + b] = p3;
}

// ---------------- host orchestration ----------------
extern "C" void kernel_launch(void* const* d_in, const int* in_sizes, int n_in,
                              void* d_out, int out_size, void* d_ws, size_t ws_size,
                              hipStream_t stream){
  (void)in_sizes; (void)n_in; (void)out_size; (void)ws_size;
  const float* features  = (const float*)d_in[0];
  const float* pnl       = (const float*)d_in[1];
  const float* embed     = (const float*)d_in[2];
  const float* res_w1    = (const float*)d_in[3];
  const float* res_w2    = (const float*)d_in[4];
  const float* pnl_w1    = (const float*)d_in[5];
  const float* pnl_w2    = (const float*)d_in[6];
  const float* stats_w   = (const float*)d_in[7];
  const float* head_len  = (const float*)d_in[8];
  const float* head_size = (const float*)d_in[9];
  const float* head_buy  = (const float*)d_in[10];
  const float* head_sell = (const float*)d_in[11];

  char* ws = (char*)d_ws;
  size_t off = 0;
  auto alloc = [&](size_t bytes) -> char* {
    char* p = ws + off;
    off = (off + bytes + 255) & ~(size_t)255;
    return p;
  };

  // bf16 weights (wk/wv zero-padded to 256 cols so 256-wide tiles need no load guards)
  u16t* emb_b = (u16t*)alloc((size_t)128 * 1024 * 2);
  u16t* rw1_b = (u16t*)alloc((size_t)1024 * 2048 * 2);
  u16t* rw2_b = (u16t*)alloc((size_t)2048 * 1024 * 2);
  u16t *wq_b[12], *wk_b[12], *wv_b[12], *wo_b[12], *fc_b[12], *pj_b[12];
  for (int l = 0; l < 12; ++l){
    wq_b[l] = (u16t*)alloc((size_t)1024 * 1024 * 2);
    wk_b[l] = (u16t*)alloc((size_t)1024 * 256 * 2);
    wv_b[l] = (u16t*)alloc((size_t)1024 * 256 * 2);
    wo_b[l] = (u16t*)alloc((size_t)1024 * 1024 * 2);
    fc_b[l] = (u16t*)alloc((size_t)1024 * 4096 * 2);
    pj_b[l] = (u16t*)alloc((size_t)4096 * 1024 * 2);
  }
  // activations
  u16t*  feat_b = (u16t*)alloc((size_t)4096 * 128 * 2);
  float* h      = (float*)alloc((size_t)4096 * 1024 * 4);
  u16t*  xb     = (u16t*)alloc((size_t)4096 * 1024 * 2);
  u16t*  mid_b  = (u16t*)alloc((size_t)4096 * 4096 * 2);
  float* qf     = (float*)alloc((size_t)4096 * 1024 * 4);
  float* kf     = (float*)alloc((size_t)4096 * 64 * 4);
  u16t*  qb     = (u16t*)alloc((size_t)4 * 16 * 1024 * 64 * 2);
  u16t*  kb     = (u16t*)alloc((size_t)4 * 1024 * 64 * 2);
  u16t*  vb     = (u16t*)alloc((size_t)4 * 1024 * 64 * 2);
  u16t*  yb     = (u16t*)alloc((size_t)4096 * 1024 * 2);
  float* cosb   = (float*)alloc((size_t)1024 * 32 * 4);
  float* sinb   = (float*)alloc((size_t)1024 * 32 * 4);
  float* t1     = (float*)alloc((size_t)4 * 1024 * 4);
  float* stats  = (float*)alloc((size_t)4 * 5 * 4);

  auto cvt = [&](const float* s, u16t* d, int n){
    int blocks = (n + 255) / 256; if (blocks > 16384) blocks = 16384;
    k_cvt_bf16<<<blocks, 256, 0, stream>>>(s, d, n);
  };
  auto cvtpad = [&](const float* s, u16t* d, int K, int Nsrc, int Ndst){
    int n = K * Ndst;
    int blocks = (n + 255) / 256; if (blocks > 16384) blocks = 16384;
    k_cvt_pad<<<blocks, 256, 0, stream>>>(s, d, K, Nsrc, Ndst);
  };
  auto gemm = [&](const u16t* A, const u16t* W, int M, int N, int K, int ldw,
                  float* Cf, u16t* Cb, const float* Res, int epi){
    dim3 g((N + 255) / 256, M / 128);
    k_gemm<<<g, 256, 0, stream>>>(A, W, M, N, K, ldw, Cf, Cb, Res, epi);
  };

  // weight / input conversion (stateless: redone each call)
  cvt(features, feat_b, 4 * 4096 * 32);
  cvt(embed, emb_b, 128 * 1024);
  cvt(res_w1, rw1_b, 1024 * 2048);
  cvt(res_w2, rw2_b, 2048 * 1024);
  for (int l = 0; l < 12; ++l){
    cvt((const float*)d_in[12 + l * 6 + 0], wq_b[l], 1024 * 1024);
    cvtpad((const float*)d_in[12 + l * 6 + 1], wk_b[l], 1024, 64, 256);
    cvtpad((const float*)d_in[12 + l * 6 + 2], wv_b[l], 1024, 64, 256);
    cvt((const float*)d_in[12 + l * 6 + 3], wo_b[l], 1024 * 1024);
    cvt((const float*)d_in[12 + l * 6 + 4], fc_b[l], 1024 * 4096);
    cvt((const float*)d_in[12 + l * 6 + 5], pj_b[l], 4096 * 1024);
  }
  k_rope_setup<<<1024, 32, 0, stream>>>(cosb, sinb);

  // stem: embed + residual gelu MLP + pnl encoder + rmsnorm
  gemm(feat_b, emb_b, 4096, 1024, 128, 1024, h, nullptr, nullptr, EPI_F32);
  cvt(h, xb, 4096 * 1024);
  gemm(xb, rw1_b, 4096, 2048, 1024, 2048, nullptr, mid_b, nullptr, EPI_GELU);
  gemm(mid_b, rw2_b, 4096, 1024, 2048, 1024, h, nullptr, h, EPI_RES);
  k_pnl_stats<<<4, 256, 0, stream>>>(pnl, stats);
  k_pnl1<<<dim3(4, 4), 256, 0, stream>>>(pnl, pnl_w1, t1);
  k_pnl2<<<dim3(4, 4), 256, 0, stream>>>(t1, pnl_w2, stats, stats_w, h);
  k_rmsn<<<4096, 256, 0, stream>>>(h, h, nullptr);   // in-place rmsn

  // transformer layers
  for (int l = 0; l < 12; ++l){
    k_rmsn<<<4096, 256, 0, stream>>>(h, nullptr, xb);
    gemm(xb, wq_b[l], 4096, 1024, 1024, 1024, qf, nullptr, nullptr, EPI_F32);
    gemm(xb, wk_b[l], 4096, 64, 1024, 256, kf, nullptr, nullptr, EPI_F32);
    gemm(xb, wv_b[l], 4096, 64, 1024, 256, nullptr, vb, nullptr, EPI_BF16);
    k_rope_q<<<dim3(1024, 16, 4), 32, 0, stream>>>(qf, qb, cosb, sinb);
    k_rope_k<<<dim3(1024, 4), 32, 0, stream>>>(kf, kb, cosb, sinb);
    k_flash<<<dim3(8, 16, 4), 256, 0, stream>>>(qb, kb, vb, yb);
    gemm(yb, wo_b[l], 4096, 1024, 1024, 1024, h, nullptr, h, EPI_RES);
    k_rmsn<<<4096, 256, 0, stream>>>(h, nullptr, xb);
    gemm(xb, fc_b[l], 4096, 4096, 1024, 4096, nullptr, mid_b, nullptr, EPI_RELU2);
    gemm(mid_b, pj_b[l], 4096, 1024, 4096, 1024, h, nullptr, h, EPI_RES);
  }

  k_heads<<<4, 256, 0, stream>>>(h, head_len, head_size, head_buy, head_sell, (float*)d_out);
}
